// ConvolutionalCapsule_22591527977119
// MI455X (gfx1250) — compile-verified, matrix operand order checked by reading, not run on hardware
//
#include <hip/hip_runtime.h>
#include <math.h>

typedef __attribute__((ext_vector_type(2)))  float    v2f;
typedef __attribute__((ext_vector_type(8)))  float    v8f;
typedef __attribute__((ext_vector_type(16))) _Float16 v16h;

#define NI    144            // 9 patch positions * 16 input capsules
#define OSZ   16
#define HP    12             // output spatial size
#define EPSF  1e-9f

// dynamic-LDS layout (floats)
#define W_OFF     0
#define W_LEN     (NI*OSZ*16)      // 36864  w[i][o][q][r]
#define POSE_OFF  (W_OFF + W_LEN)
#define POSE_LEN  (4*NI*16)        // 9216   pose[n4][i][p*4+q]
#define ACT_OFF   (POSE_OFF + POSE_LEN)
#define ACT_LEN   (4*NI)           // 576
#define RR_OFF    (ACT_OFF + ACT_LEN)
#define RR_LEN    (4*NI*OSZ)       // 9216   rr'[n4][i][o] = rr*act (also E-step logits)
#define S0_OFF    (RR_OFF + RR_LEN)
#define S0_LEN    64               // [n4][o]
#define S1_OFF    (S0_OFF + S0_LEN)
#define S1_LEN    1024             // [n4][o][pr]
#define S2_OFF    (S1_OFF + S1_LEN)
#define S2_LEN    1024
#define MEAN_OFF  (S2_OFF + S2_LEN)
#define MEAN_LEN  1024
#define I2V_OFF   (MEAN_OFF + MEAN_LEN)
#define I2V_LEN   1024             // 1/(2*var+eps)
#define CNO_OFF   (I2V_OFF + I2V_LEN)
#define CNO_LEN   64               // log(act+eps) - sum_pr log(std+eps)
#define ACTO_OFF  (CNO_OFF + CNO_LEN)
#define ACTO_LEN  64
#define LDS_FLOATS (ACTO_OFF + ACTO_LEN)   // 60160 floats = 240640 B

// votes[n,i,o,p,r] = sum_q pose[n,i,p,q] * w[i,o,q,r]
// One WMMA f32 16x16x4 per (i, o-block): M=(n4,p), N=(o4,r), K=q.
// A layout (32-bit 16x4): lanes 0-15 -> M, vreg{0,1}=K{0,1}; lanes 16-31 -> K{2,3}.
// B mirrored; D: vreg v -> M = v + (lane<16 ? 0 : 8), col = lane&15.
__device__ __forceinline__ v8f compute_votes(const float* pose_s, const float* w_s,
                                             int i, int lane, int ob)
{
    const int h  = lane & 15;
    const int hi = h >> 2;     // A: n_sub ; B: o_sub
    const int lo = h & 3;      // A: p     ; B: r
#if __has_builtin(__builtin_amdgcn_wmma_f32_16x16x4_f32)
    const int q0 = (lane < 16) ? 0 : 2;
    const int o  = ob * 4 + hi;
    v2f a, b;
    a.x = pose_s[((hi * NI) + i) * 16 + lo * 4 + q0];
    a.y = pose_s[((hi * NI) + i) * 16 + lo * 4 + q0 + 1];
    b.x = w_s[((i * OSZ + o) * 4 + q0) * 4 + lo];
    b.y = w_s[((i * OSZ + o) * 4 + q0 + 1) * 4 + lo];
    v8f c = {};
    return __builtin_amdgcn_wmma_f32_16x16x4_f32(false, a, false, b,
                                                 (short)0, c, false, false);
#else
    // K-padded f16 fallback: q in K=0..3, rest zero.
    const int o = ob * 4 + hi;
    v16h a = {};
    v16h b = {};
    if (lane < 16) {
#pragma unroll
        for (int q = 0; q < 4; ++q) {
            a[q] = (_Float16)pose_s[((hi * NI) + i) * 16 + lo * 4 + q];
            b[q] = (_Float16)w_s[((i * OSZ + o) * 4 + q) * 4 + lo];
        }
    }
    v8f c = {};
    return __builtin_amdgcn_wmma_f32_16x16x32_f16(false, a, false, b,
                                                  (short)0, c, false, false);
#endif
}

extern "C" __global__ void __launch_bounds__(256)
capsule_em_kernel(const float* __restrict__ pose_in,   // [8,14,14,256]
                  const float* __restrict__ act_in,    // [8,14,14,16]
                  const float* __restrict__ w,         // [144,16,4,4]
                  const float* __restrict__ beta_v,    // [16]
                  const float* __restrict__ beta_a,    // [16]
                  float* __restrict__ out)             // pose 294912 | act 18432
{
    extern __shared__ float smem[];
    float* w_s    = smem + W_OFF;
    float* pose_s = smem + POSE_OFF;
    float* act_s  = smem + ACT_OFF;
    float* rr_s   = smem + RR_OFF;       // holds rr' = rr * act
    float* S0     = smem + S0_OFF;
    float* S1     = smem + S1_OFF;
    float* S2     = smem + S2_OFF;
    float* mean_s = smem + MEAN_OFF;
    float* i2v_s  = smem + I2V_OFF;
    float* cno_s  = smem + CNO_OFF;
    float* acto_s = smem + ACTO_OFF;

    const int tid  = threadIdx.x;
    const int lane = tid & 31;
    // wave-uniform values -> SGPRs so the WMMA loops are uniform scalar loops
    const int wv   = __builtin_amdgcn_readfirstlane(tid >> 5);

    const int g  = blockIdx.x;           // 0..287, 4 spatial positions each
    const int n0 = g * 4;
    const int b  = n0 / (HP * HP);
    const int yy = (n0 / HP) % HP;
    const int x0 = n0 % HP;              // multiple of 4

    __builtin_prefetch(w, 0, 3);

    // stage weights
    for (int idx = tid; idx < W_LEN; idx += 256) w_s[idx] = w[idx];

    // im2col: i = (ki*3+kj)*16 + icap ; pose_s[ns][i][p*4+q]
    for (int idx = tid; idx < POSE_LEN; idx += 256) {
        int ns = idx / (NI * 16);
        int i  = (idx / 16) % NI;
        int pq = idx & 15;
        int kk = i >> 4, icap = i & 15;
        int ki = kk / 3, kj = kk % 3;
        pose_s[idx] = pose_in[((b * 14 + (yy + ki)) * 14 + (x0 + ns + kj)) * 256
                              + icap * 16 + pq];
    }
    for (int idx = tid; idx < ACT_LEN; idx += 256) {
        int ns = idx / NI, i = idx % NI;
        int kk = i >> 4, icap = i & 15;
        int ki = kk / 3, kj = kk % 3;
        act_s[idx] = act_in[((b * 14 + (yy + ki)) * 14 + (x0 + ns + kj)) * 16 + icap];
    }
    __syncthreads();
    // init rr' = act / O
    for (int idx = tid; idx < RR_LEN; idx += 256)
        rr_s[idx] = act_s[idx >> 4] * (1.0f / OSZ);
    __syncthreads();

    // per-wave WMMA work split: o-block and i-half (all SGPR)
    const int ob    = wv & 3;
    const int i_beg = (wv >> 2) * (NI / 2);
    const int i_end = i_beg + (NI / 2);
    const int col   = lane & 15;
    const int oo    = ob * 4 + (col >> 2);
    const int rcol  = col & 3;

    for (int it = 0; it < 3; ++it) {
        const float inv_temp = 1.0f + (float)it;     // 1,2,3

        // zero S0|S1|S2 (contiguous)
        for (int idx = tid; idx < S0_LEN + S1_LEN + S2_LEN; idx += 256) S0[idx] = 0.f;
        __syncthreads();

        // S0[n,o] = sum_i rr'  (cheap scalar pass)
        for (int t = tid; t < 64; t += 256) {
            int ns = t >> 4, o = t & 15;
            float s = 0.f;
            for (int i = 0; i < NI; ++i)
                s += rr_s[(ns * NI + i) * OSZ + o];
            S0[t] = s;
        }

        // ---- M-step vote pass (WMMA): S1 += rr'*v, S2 += rr'*v^2 ----
        float acc1[8], acc2[8];
#pragma unroll
        for (int v = 0; v < 8; ++v) { acc1[v] = 0.f; acc2[v] = 0.f; }
        for (int i = i_beg; i < i_end; ++i) {
            v8f d = compute_votes(pose_s, w_s, i, lane, ob);
#pragma unroll
            for (int v = 0; v < 8; ++v) {
                int M  = v + ((lane < 16) ? 0 : 8);
                int ns = M >> 2;
                float rrp = rr_s[(ns * NI + i) * OSZ + oo];
                float val = d[v];
                acc1[v] += rrp * val;
                acc2[v] += rrp * val * val;
            }
        }
#pragma unroll
        for (int v = 0; v < 8; ++v) {
            int M  = v + ((lane < 16) ? 0 : 8);
            int ns = M >> 2, p = M & 3;
            int idx = (ns * OSZ + oo) * 16 + p * 4 + rcol;
            atomicAdd(&S1[idx], acc1[v]);       // ds_add_f32
            atomicAdd(&S2[idx], acc2[v]);
        }
        __syncthreads();

        // ---- stats per (n,o): mean, 1/(2var+eps), activation ----
        for (int t = tid; t < 64; t += 256) {
            int o = t & 15;
            float s0 = S0[t];
            float denom = s0 + EPSF;
            float L = 0.f;
            for (int pr = 0; pr < 16; ++pr) {
                int idx = t * 16 + pr;
                float s1 = S1[idx], s2 = S2[idx];
                float m   = s1 / denom;
                float var = fmaxf((s2 - 2.f * m * s1 + m * m * s0) / denom, 0.f);
                mean_s[idx] = m;
                i2v_s[idx]  = 1.0f / (2.f * var + EPSF);
                L += logf(sqrtf(var) + EPSF);
            }
            float cost = s0 * (16.f * beta_v[o] + L);
            float a = 1.f / (1.f + expf(-inv_temp * (beta_a[o] - cost)));
            acto_s[t] = a;
            cno_s[t]  = logf(a + EPSF) - L;     // log(act)+o_p constant part
        }
        __syncthreads();

        if (it < 2) {
            // ---- E-step vote pass (WMMA): logits -> rr_s, softmax, * act ----
            for (int i = i_beg; i < i_end; ++i) {
                v8f d = compute_votes(pose_s, w_s, i, lane, ob);
                float slo = 0.f, shi = 0.f;
#pragma unroll
                for (int v = 0; v < 8; ++v) {
                    int M  = v + ((lane < 16) ? 0 : 8);
                    int ns = M >> 2, p = M & 3;
                    int idx = (ns * OSZ + oo) * 16 + p * 4 + rcol;
                    float t = d[v] - mean_s[idx];
                    float c = t * t * i2v_s[idx];
                    if (v < 4) slo += c; else shi += c;
                }
                // reduce over r = 4 adjacent lanes (same o, same half)
                slo += __shfl_xor(slo, 1, 32);  slo += __shfl_xor(slo, 2, 32);
                shi += __shfl_xor(shi, 1, 32);  shi += __shfl_xor(shi, 2, 32);
                if (rcol == 0) {
                    int nsl = (lane < 16) ? 0 : 2;
                    rr_s[((nsl    ) * NI + i) * OSZ + oo] = cno_s[(nsl    ) * 16 + oo] - slo;
                    rr_s[((nsl + 1) * NI + i) * OSZ + oo] = cno_s[(nsl + 1) * 16 + oo] - shi;
                }
            }
            __syncthreads();
            // softmax over o, then fold act back in: rr' = softmax(zz) * act
            for (int t = tid; t < 4 * NI; t += 256) {
                float* row = &rr_s[t * OSZ];
                float av = act_s[t];
                float mx = row[0];
                for (int o = 1; o < OSZ; ++o) mx = fmaxf(mx, row[o]);
                float s = 0.f, e[OSZ];
                for (int o = 0; o < OSZ; ++o) { e[o] = expf(row[o] - mx); s += e[o]; }
                float inv = av / s;
                for (int o = 0; o < OSZ; ++o) row[o] = e[o] * inv;
            }
            __syncthreads();
        }
    }

    // outputs: pose [n][o][p][r] then act [n][o]
    for (int t = tid; t < 4 * 256; t += 256) {
        int ns = t >> 8;
        out[(size_t)(n0 + ns) * 256 + (t & 255)] = mean_s[t];
    }
    for (int t = tid; t < 64; t += 256) {
        int ns = t >> 4, o = t & 15;
        out[294912 + (size_t)(n0 + ns) * 16 + o] = acto_s[t];
    }
}

extern "C" void kernel_launch(void* const* d_in, const int* in_sizes, int n_in,
                              void* d_out, int out_size, void* d_ws, size_t ws_size,
                              hipStream_t stream) {
    const float* pose = (const float*)d_in[0];
    const float* act  = (const float*)d_in[1];
    const float* w    = (const float*)d_in[2];
    const float* bv   = (const float*)d_in[3];
    const float* ba   = (const float*)d_in[4];
    float* out = (float*)d_out;
    (void)in_sizes; (void)n_in; (void)out_size; (void)d_ws; (void)ws_size;

    const size_t lds_bytes = (size_t)LDS_FLOATS * sizeof(float);  // ~235 KB < 320 KB/WGP
    hipLaunchKernelGGL(capsule_em_kernel, dim3(288), dim3(256), lds_bytes, stream,
                       pose, act, w, bv, ba, out);
}
